// MoEFFN_41120016892131
// MI455X (gfx1250) — matrix operand order repaired and checked by reading, not compiled, not run on hardware
//
#include <hip/hip_runtime.h>
#include <hip/hip_bf16.h>
#include <math.h>

#define BN 4096      // N = B*T tokens
#define DD 1024      // model dim
#define FF 4096      // ffn dim
#define EE 8         // experts
#define EPS_INV 20.0f   // 1/0.05
#define SK_ITERS 20

typedef __bf16 bf16_t;
typedef __attribute__((ext_vector_type(16))) __bf16 v16bf;
typedef __attribute__((ext_vector_type(8)))  __bf16 v8bf;
typedef __attribute__((ext_vector_type(8)))  float  v8f;

// ---- WMMA helper: D = A(16x32 bf16) * B(32x16 bf16) + C(16x16 f32) ----
__device__ __forceinline__ v8f wmma_bf16(v16bf a, v16bf b, v8f c) {
  return __builtin_amdgcn_wmma_f32_16x16x32_bf16(false, a, false, b, (short)0, c,
                                                 false, false);
}

// B-fragment: elems 0..7 = p[0..7], elems 8..15 = p[16..23], convert to bf16.
__device__ __forceinline__ v16bf load_cvt_b(const float* __restrict__ p) {
  const float4* q = (const float4*)p;
  float4 f0 = q[0], f1 = q[1], f2 = q[4], f3 = q[5];
  v16bf r;
  r[0]=(bf16_t)f0.x;  r[1]=(bf16_t)f0.y;  r[2]=(bf16_t)f0.z;  r[3]=(bf16_t)f0.w;
  r[4]=(bf16_t)f1.x;  r[5]=(bf16_t)f1.y;  r[6]=(bf16_t)f1.z;  r[7]=(bf16_t)f1.w;
  r[8]=(bf16_t)f2.x;  r[9]=(bf16_t)f2.y;  r[10]=(bf16_t)f2.z; r[11]=(bf16_t)f2.w;
  r[12]=(bf16_t)f3.x; r[13]=(bf16_t)f3.y; r[14]=(bf16_t)f3.z; r[15]=(bf16_t)f3.w;
  return r;
}

// A-fragment from fp32 row: elems 0..7 = p[0..7], elems 8..15 = p[16..23]
// (ISA 16-bit A layout: per lane, K = c0..c0+7 and c0+16..c0+23, c0 = 8*(lane>>4))
__device__ __forceinline__ v16bf load_cvt_a(const float* __restrict__ p) {
  const float4* q = (const float4*)p;
  float4 f0 = q[0], f1 = q[1], f2 = q[4], f3 = q[5];
  v16bf r;
  r[0]=(bf16_t)f0.x;  r[1]=(bf16_t)f0.y;  r[2]=(bf16_t)f0.z;  r[3]=(bf16_t)f0.w;
  r[4]=(bf16_t)f1.x;  r[5]=(bf16_t)f1.y;  r[6]=(bf16_t)f1.z;  r[7]=(bf16_t)f1.w;
  r[8]=(bf16_t)f2.x;  r[9]=(bf16_t)f2.y;  r[10]=(bf16_t)f2.z; r[11]=(bf16_t)f2.w;
  r[12]=(bf16_t)f3.x; r[13]=(bf16_t)f3.y; r[14]=(bf16_t)f3.z; r[15]=(bf16_t)f3.w;
  return r;
}

// A-fragment from bf16 row (h buffer)
__device__ __forceinline__ v16bf load_a_bf16(const bf16_t* __restrict__ p) {
  v8bf lo = *(const v8bf*)(p);
  v8bf hi = *(const v8bf*)(p + 16);
  return __builtin_shufflevector(lo, hi, 0,1,2,3,4,5,6,7,8,9,10,11,12,13,14,15);
}

// ---------------- zero-init output ----------------
__global__ __launch_bounds__(256) void zero_f32_kernel(float* __restrict__ p, int n4) {
  int i = blockIdx.x * 256 + threadIdx.x;
  if (i < n4) ((float4*)p)[i] = make_float4(0.f, 0.f, 0.f, 0.f);
}

// ---------------- gating logits: (N,E) = x @ gate_W^T ----------------
__global__ __launch_bounds__(256) void gate_logits_kernel(
    const float* __restrict__ x, const float* __restrict__ gW,
    float* __restrict__ logits) {
  int idx = blockIdx.x * 256 + threadIdx.x;   // grid covers N*E exactly
  int n = idx >> 3, e = idx & 7;
  const float4* xr = (const float4*)(x + (size_t)n * DD);
  const float4* wr = (const float4*)(gW + (size_t)e * DD);
  float s = 0.f;
  #pragma unroll 8
  for (int d = 0; d < DD / 4; ++d) {
    float4 a = xr[d], b = wr[d];
    s += a.x * b.x + a.y * b.y + a.z * b.z + a.w * b.w;
  }
  logits[n * EE + e] = s;
}

// -------- Sinkhorn (log-domain) + top-2 routing, single workgroup --------
__global__ __launch_bounds__(1024) void sinkhorn_route_kernel(
    const float* __restrict__ logits, int* __restrict__ counts,
    int* __restrict__ base, int* __restrict__ tok_list,
    float* __restrict__ wgt_list) {
  __shared__ float red[EE * 1024];
  __shared__ int scnt[EE];
  const int tid = threadIdx.x;
  float la[4][EE];                       // 4 tokens per thread
  #pragma unroll
  for (int r = 0; r < 4; ++r) {
    int n = tid * 4 + r;
    #pragma unroll
    for (int e = 0; e < EE; ++e) la[r][e] = -logits[n * EE + e] * EPS_INV;
  }
  for (int it = 0; it < SK_ITERS; ++it) {
    // row logsumexp over E=8 (local, max-stabilized)
    #pragma unroll
    for (int r = 0; r < 4; ++r) {
      float m = la[r][0];
      #pragma unroll
      for (int e = 1; e < EE; ++e) m = fmaxf(m, la[r][e]);
      float s = 0.f;
      #pragma unroll
      for (int e = 0; e < EE; ++e) s += __expf(la[r][e] - m);
      float lse = m + __logf(s);
      #pragma unroll
      for (int e = 0; e < EE; ++e) la[r][e] -= lse;
    }
    // column logsumexp over N=4096 per expert; entries <= 0 so plain sum-exp is safe
    float part[EE];
    #pragma unroll
    for (int e = 0; e < EE; ++e) {
      float s = 0.f;
      #pragma unroll
      for (int r = 0; r < 4; ++r) s += __expf(la[r][e]);
      part[e] = s;
    }
    __syncthreads();
    #pragma unroll
    for (int e = 0; e < EE; ++e) red[e * 1024 + tid] = part[e];
    __syncthreads();
    for (int s2 = 512; s2 > 0; s2 >>= 1) {
      if (tid < s2) {
        #pragma unroll
        for (int e = 0; e < EE; ++e)
          red[e * 1024 + tid] += red[e * 1024 + tid + s2];
      }
      __syncthreads();
    }
    #pragma unroll
    for (int e = 0; e < EE; ++e) {
      float lse = __logf(red[e * 1024]);
      #pragma unroll
      for (int r = 0; r < 4; ++r) la[r][e] -= lse;
    }
  }
  // top-2 + per-expert compacted lists
  if (tid < EE) scnt[tid] = 0;
  __syncthreads();
  #pragma unroll
  for (int r = 0; r < 4; ++r) {
    int n = tid * 4 + r;
    float p[EE];
    #pragma unroll
    for (int e = 0; e < EE; ++e) p[e] = __expf(la[r][e]);
    int i0 = 0; float v0 = p[0];
    #pragma unroll
    for (int e = 1; e < EE; ++e) if (p[e] > v0) { v0 = p[e]; i0 = e; }
    int i1 = (i0 == 0) ? 1 : 0; float v1 = p[i1];
    #pragma unroll
    for (int e = 0; e < EE; ++e)
      if (e != i0 && p[e] > v1) { v1 = p[e]; i1 = e; }
    // bug-faithful combine weights: columns 0 and 1 of the transport plan pi
    int pos0 = atomicAdd(&scnt[i0], 1);
    tok_list[i0 * BN + pos0] = n;
    wgt_list[i0 * BN + pos0] = p[0];
    int pos1 = atomicAdd(&scnt[i1], 1);
    tok_list[i1 * BN + pos1] = n;
    wgt_list[i1 * BN + pos1] = p[1];
  }
  __syncthreads();
  if (tid == 0) {
    int acc = 0;
    for (int e = 0; e < EE; ++e) { counts[e] = scnt[e]; base[e] = acc; acc += scnt[e]; }
  }
}

// -------- expert gate/up GEMM + SiLU fusion: h = silu(xWg^T) * (xWu^T) --------
// block = 8 waves, tile M128 x F128; each wave: M64 x F32 for both g and u.
// Each B fragment feeds 4 WMMAs (a0..a3) -> WMMA->VALU WAR hazard amortized
// over 4 WMMAs and B loads/cvts per WMMA halved again.
__global__ __launch_bounds__(256) void expert_gateup_kernel(
    const float* __restrict__ x, const float* __restrict__ Wg,
    const float* __restrict__ Wu, const int* __restrict__ counts,
    const int* __restrict__ base, const int* __restrict__ tok_list,
    bf16_t* __restrict__ hbuf) {
  const int e = blockIdx.z;
  const int cnt = counts[e];
  const int m0 = blockIdx.y * 128;
  if (m0 >= cnt) return;
  const int f0 = blockIdx.x * 128;
  const int tid = threadIdx.x;
  const int w = tid >> 5, lane = tid & 31;
  const int mi = w & 1, fj = w >> 1;      // 2 M-groups x 4 F-groups
  const int mrow = m0 + mi * 64;          // wave owns M64
  const int fbase = f0 + fj * 32;         // wave owns F32
  const int lm = lane & 15, kh = lane >> 4;
  const int c0 = kh * 8;

  const float* xr[4];
  #pragma unroll
  for (int t = 0; t < 4; ++t) {
    int r = mrow + t * 16 + lm;
    int tk = (r < cnt) ? tok_list[e * BN + r] : 0;
    xr[t] = x + (size_t)tk * DD;
  }
  const float* wgp = Wg + (size_t)e * FF * DD;   // uniform expert base (SGPR)
  const float* wup = Wu + (size_t)e * FF * DD;
  const unsigned bcol = (unsigned)(fbase + lm) * DD + (unsigned)(kh * 8);

  v8f accg[4][2] = {}; v8f accu[4][2] = {};
  for (int k0 = 0; k0 < DD; k0 += 32) {
    v16bf a0 = load_cvt_a(xr[0] + k0 + c0);
    v16bf a1 = load_cvt_a(xr[1] + k0 + c0);
    v16bf a2 = load_cvt_a(xr[2] + k0 + c0);
    v16bf a3 = load_cvt_a(xr[3] + k0 + c0);
    #pragma unroll
    for (int s = 0; s < 2; ++s) {
      unsigned o = bcol + (unsigned)(s * 16) * DD + (unsigned)k0;
      v16bf bg = load_cvt_b(wgp + o);
      accg[0][s] = wmma_bf16(a0, bg, accg[0][s]);
      accg[1][s] = wmma_bf16(a1, bg, accg[1][s]);
      accg[2][s] = wmma_bf16(a2, bg, accg[2][s]);
      accg[3][s] = wmma_bf16(a3, bg, accg[3][s]);
      v16bf bu = load_cvt_b(wup + o);
      accu[0][s] = wmma_bf16(a0, bu, accu[0][s]);
      accu[1][s] = wmma_bf16(a1, bu, accu[1][s]);
      accu[2][s] = wmma_bf16(a2, bu, accu[2][s]);
      accu[3][s] = wmma_bf16(a3, bu, accu[3][s]);
    }
  }
  const size_t rowbase = (size_t)base[e];
  #pragma unroll
  for (int t = 0; t < 4; ++t) {
    #pragma unroll
    for (int s = 0; s < 2; ++s) {
      int fcol = fbase + s * 16 + lm;
      #pragma unroll
      for (int rr = 0; rr < 8; ++rr) {     // C layout: m = rr + 8*(lane>>4)
        int mloc = mrow + t * 16 + rr + 8 * kh;
        if (mloc < cnt) {
          float g = accg[t][s][rr], u = accu[t][s][rr];
          float hv = (g / (1.f + __expf(-g))) * u;   // silu(g)*u
          hbuf[(rowbase + (size_t)mloc) * FF + fcol] = (bf16_t)hv;
        }
      }
    }
  }
}

// -------- expert down GEMM + weighted scatter: out[tok] += w * (h Wd^T) --------
// block = 8 waves, tile M128 x D128; each wave: M64 x D32.
__global__ __launch_bounds__(256) void expert_down_kernel(
    const bf16_t* __restrict__ hbuf, const float* __restrict__ Wd,
    const int* __restrict__ counts, const int* __restrict__ base,
    const int* __restrict__ tok_list, const float* __restrict__ wgt_list,
    float* __restrict__ out) {
  const int e = blockIdx.z;
  const int cnt = counts[e];
  const int m0 = blockIdx.y * 128;
  if (m0 >= cnt) return;
  const int d0 = blockIdx.x * 128;
  const int tid = threadIdx.x;
  const int w = tid >> 5, lane = tid & 31;
  const int mi = w & 1, dj = w >> 1;      // 2 M-groups x 4 D-groups
  const int mrow = m0 + mi * 64;
  const int dbase = d0 + dj * 32;
  const int lm = lane & 15, kh = lane >> 4;
  const int c0 = kh * 8;

  const bf16_t* hr[4];
  #pragma unroll
  for (int t = 0; t < 4; ++t) {
    int r = mrow + t * 16 + lm;
    int rc = (r < cnt) ? r : (cnt - 1);
    hr[t] = hbuf + ((size_t)base[e] + (size_t)rc) * FF;
  }
  const float* wdp = Wd + (size_t)e * DD * FF;   // uniform expert base
  const unsigned bcol = (unsigned)(dbase + lm) * FF + (unsigned)(kh * 8);

  v8f acc[4][2] = {};
  for (int k0 = 0; k0 < FF; k0 += 32) {
    v16bf a0 = load_a_bf16(hr[0] + k0 + c0);
    v16bf a1 = load_a_bf16(hr[1] + k0 + c0);
    v16bf a2 = load_a_bf16(hr[2] + k0 + c0);
    v16bf a3 = load_a_bf16(hr[3] + k0 + c0);
    #pragma unroll
    for (int s = 0; s < 2; ++s) {
      v16bf b = load_cvt_b(wdp + bcol + (unsigned)(s * 16) * FF + (unsigned)k0);
      acc[0][s] = wmma_bf16(a0, b, acc[0][s]);
      acc[1][s] = wmma_bf16(a1, b, acc[1][s]);
      acc[2][s] = wmma_bf16(a2, b, acc[2][s]);
      acc[3][s] = wmma_bf16(a3, b, acc[3][s]);
    }
  }
  #pragma unroll
  for (int t = 0; t < 4; ++t) {
    #pragma unroll
    for (int rr = 0; rr < 8; ++rr) {
      int mloc = mrow + t * 16 + rr + 8 * kh;
      if (mloc < cnt) {
        int   tok = tok_list[e * BN + mloc];
        float wv  = wgt_list[e * BN + mloc];
        float* orow = out + (size_t)tok * DD;
        #pragma unroll
        for (int s = 0; s < 2; ++s)
          unsafeAtomicAdd(orow + dbase + s * 16 + lm, wv * acc[t][s][rr]);
      }
    }
  }
}

extern "C" void kernel_launch(void* const* d_in, const int* in_sizes, int n_in,
                              void* d_out, int out_size, void* d_ws, size_t ws_size,
                              hipStream_t stream) {
  const float* x  = (const float*)d_in[0];   // (B,T,D) fp32
  const float* gW = (const float*)d_in[1];   // (E,D)
  const float* Wg = (const float*)d_in[2];   // (E,F,D)
  const float* Wu = (const float*)d_in[3];   // (E,F,D)
  const float* Wd = (const float*)d_in[4];   // (E,D,F)
  float* out = (float*)d_out;                // (B,T,D) fp32

  // workspace layout (~68 MB total)
  char* ws = (char*)d_ws;
  float*  logits = (float*)(ws + 0);          // N*E*4      = 131072 B
  int*    counts = (int*)  (ws + 131072);     // E*4 (padded to 256)
  int*    basep  = (int*)  (ws + 131328);     // E*4 (padded to 256)
  int*    tok    = (int*)  (ws + 131584);     // E*N*4      = 131072 B
  float*  wgt    = (float*)(ws + 262656);     // E*N*4      = 131072 B
  bf16_t* hbuf   = (bf16_t*)(ws + 393728);    // 2N*F*2     = 64 MB

  zero_f32_kernel<<<(BN * DD / 4 + 255) / 256, 256, 0, stream>>>(out, BN * DD / 4);
  gate_logits_kernel<<<BN * EE / 256, 256, 0, stream>>>(x, gW, logits);
  sinkhorn_route_kernel<<<1, 1024, 0, stream>>>(logits, counts, basep, tok, wgt);
  expert_gateup_kernel<<<dim3(FF / 128, BN / 128, EE), 256, 0, stream>>>(
      x, Wg, Wu, counts, basep, tok, hbuf);
  expert_down_kernel<<<dim3(DD / 128, BN / 128, EE), 256, 0, stream>>>(
      hbuf, Wd, counts, basep, tok, wgt, out);
}
